// TuckerConv_89137751261665
// MI455X (gfx1250) — matrix operand order repaired and checked by reading, not compiled, not run on hardware
//
#include <hip/hip_runtime.h>
#include <hip/hip_bf16.h>
#include <stdint.h>

typedef __bf16 bf16_t;
typedef bf16_t v16bf __attribute__((ext_vector_type(16)));
typedef float  v8f   __attribute__((ext_vector_type(8)));

#define KDIM     1024
#define KSTRIDE  1032   // bf16 units: 1024 + 8 pad -> 2064 B row = 516 dwords (== 4 mod 64 banks)
#define NPIX     64     // pixels per block: 4 rows x 16 cols
#define LDS_BYTES (NPIX * KSTRIDE * 2)   // 132096 B

static __device__ __forceinline__ unsigned short f2bf(float f) {
  unsigned int u = __float_as_uint(f);
  u += 0x7FFFu + ((u >> 16) & 1u);   // round-to-nearest-even
  return (unsigned short)(u >> 16);
}

union ABPack { uint4 q[2]; v16bf v; };

// -------- Kernel 1: w_eff[m=512][k=1024] in bf16 (row-major, K contiguous) --------
// m = x*32 + y ; k = (i*16 + j)*4 + s ; w_eff = sum_{r,s2} f0[i,r,x] f1[j,s2,y] core[r,s2,s]
__global__ __launch_bounds__(256) void tucker_weff(
    const float* __restrict__ f0, const float* __restrict__ f1,
    const float* __restrict__ core, unsigned short* __restrict__ w2) {
  int e  = blockIdx.x * 256 + threadIdx.x;   // 0 .. 512*1024-1
  int m  = e >> 10, k = e & 1023;
  int xo = m >> 5,  yo = m & 31;
  int c  = k >> 2,  s  = k & 3;
  int i  = c >> 4,  j  = c & 15;
  float acc = 0.f;
#pragma unroll
  for (int r = 0; r < 8; ++r) {
    float a = f0[i * 128 + r * 16 + xo];
#pragma unroll
    for (int s2 = 0; s2 < 8; ++s2)
      acc += a * f1[j * 256 + s2 * 32 + yo] * core[r * 32 + s2 * 4 + s];
  }
  w2[e] = f2bf(acc);
}

// -------- Kernel 2: fused depthwise conv (f32) + channel-mix GEMM (bf16 WMMA) --------
// Block = 64 pixels (4 rows x 16 cols) of one image. 256 threads = 8 waves.
// Each wave: 4 M-tiles (64 out-channels) x 4 N-tiles (all 64 pixels).
__global__ __launch_bounds__(256) void tucker_main(
    const float* __restrict__ x, const float* __restrict__ sf,
    const unsigned short* __restrict__ w2, const float* __restrict__ bias,
    float* __restrict__ out) {
  extern __shared__ unsigned short tLds[];   // [64 pixels][KSTRIDE] bf16
  const int tid = threadIdx.x;
  const int blk = blockIdx.x;        // 0..1023
  const int b   = blk >> 6;          // image (16)
  const int rem = blk & 63;
  const int r0  = (rem >> 2) << 2;   // row-quad base (16 quads)
  const int c0  = (rem & 3) << 4;    // col-tile base (4 tiles)

  // Spatial factors: uniform addresses -> scalar loads
  float sfr[36];
#pragma unroll
  for (int z = 0; z < 36; ++z) sfr[z] = sf[z];

  // ---- Phase 1: depthwise 3x3 conv -> t tile in LDS (bf16), layout [n][k], k = c*4+s ----
#pragma unroll 1
  for (int it = 0; it < 64; ++it) {
    int idx = it * 256 + tid;          // 16384 (c, n) pairs
    int c = idx >> 6, n = idx & 63;
    int h  = r0 + (n >> 4);
    int wv = c0 + (n & 15);
    const float* xb = x + ((size_t)((b << 8) + c) << 12);  // x[b][c][.][.]
    float a0 = 0.f, a1 = 0.f, a2 = 0.f, a3 = 0.f;
#pragma unroll
    for (int dh = -1; dh <= 1; ++dh) {
      int hh = h + dh;
      bool hin = (hh >= 0) & (hh < 64);
#pragma unroll
      for (int dw = -1; dw <= 1; ++dw) {
        int ww = wv + dw;
        bool win = hin & (ww >= 0) & (ww < 64);
        float v = win ? xb[hh * 64 + ww] : 0.f;
        int tap = (dh + 1) * 3 + (dw + 1);
        a0 += v * sfr[tap];
        a1 += v * sfr[9  + tap];
        a2 += v * sfr[18 + tap];
        a3 += v * sfr[27 + tap];
      }
    }
    uint2 pk;
    pk.x = (unsigned)f2bf(a0) | ((unsigned)f2bf(a1) << 16);
    pk.y = (unsigned)f2bf(a2) | ((unsigned)f2bf(a3) << 16);
    *(uint2*)(&tLds[n * KSTRIDE + (c << 2)]) = pk;   // 8B-aligned ds_store_b64
  }
  __syncthreads();

  // ---- Phase 2: GEMM  out[512 x 64] += w_eff^T[512 x 1024] * t[1024 x 64] ----
  const int lane = tid & 31;
  const int wid  = tid >> 5;        // 8 waves x 4 M-tiles = 32 M-tiles (512 channels)
  const int ml   = lane & 15;       // A: row within M-tile / D: column
  const int half = lane >> 4;       // A/B K-half select

  const v8f vzero = {0.f, 0.f, 0.f, 0.f, 0.f, 0.f, 0.f, 0.f};
  v8f acc[4][4];                    // [Mtile][Ntile]
#pragma unroll
  for (int q = 0; q < 4; ++q)
#pragma unroll
    for (int nt = 0; nt < 4; ++nt) acc[q][nt] = vzero;

  const unsigned short* bBase = &tLds[ml * KSTRIDE + (half << 3)];  // + khalf*8
  const unsigned short* aBase = w2 + ((((wid << 2) << 4) + ml) * KDIM) + (half << 3);

#pragma unroll 1
  for (int kk = 0; kk < 32; ++kk) {
    int kb = kk << 5;
    // A 16x32 bf16 layout: VGPR0-3 -> K = h*8 + 0..7 ; VGPR4-7 -> K = 16 + h*8 + 0..7
    ABPack A[4];
#pragma unroll
    for (int q = 0; q < 4; ++q) {
      const unsigned short* ap = aBase + (q << 4) * KDIM + kb;
      A[q].q[0] = *(const uint4*)(ap);
      A[q].q[1] = *(const uint4*)(ap + 16);
    }
#pragma unroll
    for (int nt = 0; nt < 4; ++nt) {
      // B 32x16 bf16 layout: VGPR0-3 -> K = h*8 + 0..7 ; VGPR4-7 -> K = 16 + h*8 + 0..7
      const unsigned short* bp = bBase + (nt << 4) * KSTRIDE + kb;
      ABPack B;
      B.q[0] = *(const uint4*)(bp);
      B.q[1] = *(const uint4*)(bp + 16);
      acc[0][nt] = __builtin_amdgcn_wmma_f32_16x16x32_bf16(
          false, A[0].v, false, B.v, (short)0, acc[0][nt], false, false);
      acc[1][nt] = __builtin_amdgcn_wmma_f32_16x16x32_bf16(
          false, A[1].v, false, B.v, (short)0, acc[1][nt], false, false);
      acc[2][nt] = __builtin_amdgcn_wmma_f32_16x16x32_bf16(
          false, A[2].v, false, B.v, (short)0, acc[2][nt], false, false);
      acc[3][nt] = __builtin_amdgcn_wmma_f32_16x16x32_bf16(
          false, A[3].v, false, B.v, (short)0, acc[3][nt], false, false);
    }
  }

  // ---- Phase 3: bias add + store. C/D layout: VGPR v -> M = v + half*8, N = lane%16 ----
#pragma unroll
  for (int q = 0; q < 4; ++q) {
    int mbase = (((wid << 2) + q) << 4) + (half << 3);
    float bv[8];
#pragma unroll
    for (int v = 0; v < 8; ++v) bv[v] = bias[mbase + v];
#pragma unroll
    for (int nt = 0; nt < 4; ++nt) {
      int row = r0 + nt;
      int col = c0 + ml;
#pragma unroll
      for (int v = 0; v < 8; ++v) {
        int m = mbase + v;
        out[((size_t)((b << 9) + m) << 12) + (row << 6) + col] = acc[q][nt][v] + bv[v];
      }
    }
  }
}

// -------- Host launch --------
extern "C" void kernel_launch(void* const* d_in, const int* in_sizes, int n_in,
                              void* d_out, int out_size, void* d_ws, size_t ws_size,
                              hipStream_t stream) {
  (void)in_sizes; (void)n_in; (void)out_size; (void)ws_size;
  const float* x    = (const float*)d_in[0];
  const float* sf   = (const float*)d_in[1];
  const float* f0   = (const float*)d_in[2];
  const float* f1   = (const float*)d_in[3];
  const float* core = (const float*)d_in[4];
  const float* bias = (const float*)d_in[5];
  unsigned short* w2 = (unsigned short*)d_ws;      // 512*1024 bf16 = 1 MB

  tucker_weff<<<2048, 256, 0, stream>>>(f0, f1, core, w2);
  tucker_main<<<1024, 256, LDS_BYTES, stream>>>(x, sf, w2, bias, (float*)d_out);
}
